// PacJointUpsample_29222957482304
// MI455X (gfx1250) — matrix operand order repaired and checked by reading, not compiled, not run on hardware
//
#include <hip/hip_runtime.h>

// ---------------------------------------------------------------------------
// PAC joint upsample, CDNA5 (gfx1250, wave32).
// Every conv is an implicit GEMM on v_wmma_f32_16x16x32_f16.
// Per-K gather descriptors + weight B-panels are staged into LDS with two
// Tensor Data Mover loads per workgroup; the A gather is branchless and
// software-pipelined (issue all 16 loads, then convert/select).
// ---------------------------------------------------------------------------

typedef __attribute__((ext_vector_type(16))) _Float16 v16h;
typedef __attribute__((ext_vector_type(2)))  _Float16 h2;
typedef __attribute__((ext_vector_type(8)))  float    v8f;
typedef __attribute__((ext_vector_type(4)))  unsigned int u32x4;
typedef __attribute__((ext_vector_type(8)))  int      i32x8;
typedef __attribute__((ext_vector_type(4)))  int      i32x4;

#define BS 2
#define HG 192
#define PG (192 * 192)
#define H0 96
#define P0 (96 * 96)
#define KBAD (1 << 29)   // poison dy for padded K slots -> always out of range

// ---------------------------- TDM staging ----------------------------------
#if defined(__AMDGCN__) && __has_builtin(__builtin_amdgcn_tensor_load_to_lds)
#define HAVE_TDM 1
#endif

#ifdef HAVE_TDM
__device__ __forceinline__ void tdm_issue(void* lds, const void* gsrc, int bytes) {
  unsigned lds_off = (unsigned)(unsigned long long)lds;   // LDS byte offset = addr[31:0]
  unsigned long long ga = (unsigned long long)gsrc;
  int ne = bytes >> 3;                                    // 8-byte elements
  u32x4 g0;
  g0[0] = 1u;                                             // count=1, user mode
  g0[1] = lds_off;                                        // lds_addr
  g0[2] = (unsigned)(ga & 0xffffffffull);                 // global_addr lo
  g0[3] = (unsigned)((ga >> 32) & 0x01ffffffull) | (2u << 30); // addr hi | type=2
  i32x8 g1;
  g1[0] = 3 << 16;                                        // data_size = 8B
  g1[1] = (ne & 0xffff) << 16;                            // tensor_dim0[15:0]
  g1[2] = ((ne >> 16) & 0xffff) | (1 << 16);              // dim0 hi | tensor_dim1=1
  g1[3] = (ne & 0xffff) << 16;                            // tile_dim0 = ne
  g1[4] = 1;                                              // tile_dim1 = 1
  g1[5] = ne;                                             // tensor_dim0_stride lo32
  g1[6] = 0;
  g1[7] = 0;
  i32x4 z4 = (i32x4)0;
#if defined(__clang_major__) && __clang_major__ >= 23
  i32x8 z8 = (i32x8)0;
  __builtin_amdgcn_tensor_load_to_lds(g0, g1, z4, z4, z8, 0);
#else
  __builtin_amdgcn_tensor_load_to_lds(g0, g1, z4, z4, 0);
#endif
}
#endif

// ---------------------------- WMMA conv ------------------------------------
enum { MK5F32 = 0, MK5F16 = 1, MGRU = 2, MPAC = 3 };

struct ConvArgs {
  const void* in0;
  const void* in1;          // second tensor (GRU hidden / r*h)
  const float* kmap;        // PAC affinity
  const _Float16* bpack;    // packed B panels: per N-tile [Kpad/2][16][2] f16
  const int4* desc;         // per-K gather descriptor {dy, dx, chanOff, aux}
  const float* bias;
  void* out;
  int H, W, P, Cout, Kchunks, stride0;  // stride0: elements per image in in0/in1
};

template <int MODE, bool RELU, bool OUTH>
__global__ void wmma_conv(ConvArgs a) {
  extern __shared__ char lds_raw[];
  const int lane = threadIdx.x & 31;
  const int wid = threadIdx.x >> 5;
  const int ntile = blockIdx.y;
  const int panelBytes = a.Kchunks << 10;  // Kchunks*32 K * 16 N * 2B
  const int descBytes = a.Kchunks << 9;    // Kchunks*32 K * 16B
  char* ldsB = lds_raw;
  char* ldsD = lds_raw + panelBytes;

#ifdef HAVE_TDM
  if (wid == 0) {  // wave 0 issues both DMAs (uniform branch)
    tdm_issue(ldsB, (const char*)a.bpack + (size_t)ntile * panelBytes, panelBytes);
    tdm_issue(ldsD, a.desc, descBytes);
    __builtin_amdgcn_s_wait_tensorcnt(0);
  }
#else
  {
    const unsigned* s = (const unsigned*)((const char*)a.bpack + (size_t)ntile * panelBytes);
    unsigned* dd = (unsigned*)ldsB;
    for (int i = threadIdx.x; i < (panelBytes >> 2); i += blockDim.x) dd[i] = s[i];
    const unsigned* s2 = (const unsigned*)a.desc;
    unsigned* d2 = (unsigned*)ldsD;
    for (int i = threadIdx.x; i < (descBytes >> 2); i += blockDim.x) d2[i] = s2[i];
  }
#endif
  __syncthreads();
  const _Float16* Bl = (const _Float16*)ldsB;
  const int4* Dl = (const int4*)ldsD;

  const int lhalf = lane >> 4;
  const int l15 = lane & 15;
  const int mTile = ((blockIdx.x << 2) + wid) << 4;  // 4 waves/WG, 16 rows each
  // P is a multiple of 16 -> a tile never straddles images; decode once.
  const int img = mTile / a.P;
  const int hwb = mTile - img * a.P;
  const int hw = hwb + l15;                          // A row for this lane
  const int y = hw / a.W;
  const int x = hw - y * a.W;
  const size_t imgOff = (size_t)img * a.stride0;

  v8f acc = (v8f)0.0f;
  for (int kc = 0; kc < a.Kchunks; ++kc) {
    v16h av, bv;
    // B fragment: VGPR v holds K = {2p, 2p+1}, p = kc*16 + v + 8*half; N = lane&15
#pragma unroll
    for (int v = 0; v < 8; ++v) {
      int p = (kc << 4) + v + (lhalf << 3);
      h2 pr = *(const h2*)(Bl + (((p << 4) + l15) << 1));
      bv[2 * v] = pr[0];
      bv[2 * v + 1] = pr[1];
    }
    // A fragment K indices: K = 2v+16*half+t
    int ks[16];
#pragma unroll
    for (int i = 0; i < 16; ++i) {
      int v = i >> 1, t = i & 1;
      int kk = 2 * v + (lhalf << 4) + t;
      ks[i] = (kc << 5) + kk;
    }
    if (MODE == MK5F16 || MODE == MGRU) {
      _Float16 raw[16];
      bool vld[16];
#pragma unroll
      for (int i = 0; i < 16; ++i) {           // phase 1: issue all loads
        int4 d = Dl[ks[i]];
        int iy = y + d.x, ix = x + d.y;
        bool val = ((unsigned)iy < (unsigned)a.H) && ((unsigned)ix < (unsigned)a.W);
        int off = d.z + iy * a.W + ix;
        off = val ? off : 0;
        const _Float16* base = (MODE == MGRU && d.w) ? (const _Float16*)a.in1
                                                     : (const _Float16*)a.in0;
        raw[i] = base[imgOff + off];
        vld[i] = val;
      }
#pragma unroll
      for (int i = 0; i < 16; ++i)             // phase 2: select
        av[i] = vld[i] ? raw[i] : (_Float16)0;
    } else if (MODE == MK5F32) {
      float raw[16];
      bool vld[16];
#pragma unroll
      for (int i = 0; i < 16; ++i) {
        int4 d = Dl[ks[i]];
        int iy = y + d.x, ix = x + d.y;
        bool val = ((unsigned)iy < (unsigned)a.H) && ((unsigned)ix < (unsigned)a.W);
        int off = d.z + iy * a.W + ix;
        off = val ? off : 0;
        raw[i] = ((const float*)a.in0)[imgOff + off];
        vld[i] = val;
      }
#pragma unroll
      for (int i = 0; i < 16; ++i)
        av[i] = (_Float16)(vld[i] ? raw[i] : 0.0f);
    } else {  // MPAC: zero-upsampled xt * affinity K folded into A
      float xv[16], kv[16];
      bool vld[16];
#pragma unroll
      for (int i = 0; i < 16; ++i) {
        int4 d = Dl[ks[i]];
        int uy = y + d.x, ux = x + d.y;
        int m = uy | ux;
        bool val = (m >= 0) && ((m & 1) == 0);
        int uy2 = uy >> 1, ux2 = ux >> 1;
        val = val && ((unsigned)uy2 < (unsigned)H0) && ((unsigned)ux2 < (unsigned)H0);
        int off = d.z + uy2 * H0 + ux2;
        off = val ? off : 0;
        xv[i] = ((const float*)a.in0)[imgOff + off];
        kv[i] = a.kmap[((size_t)(img * 25 + d.w)) * a.P + hw];
        vld[i] = val;
      }
#pragma unroll
      for (int i = 0; i < 16; ++i)
        av[i] = vld[i] ? (_Float16)(xv[i] * kv[i]) : (_Float16)0;
    }
    acc = __builtin_amdgcn_wmma_f32_16x16x32_f16(false, av, false, bv,
                                                 (short)0, acc, false, false);
  }

  // C/D layout: N = lane&15, M = r + 8*half
  const int ch = (ntile << 4) + l15;
  const float bval = (ch < a.Cout) ? a.bias[ch] : 0.0f;
#pragma unroll
  for (int r = 0; r < 8; ++r) {
    if (ch < a.Cout) {
      int ohw = hwb + r + (lhalf << 3);
      float v = acc[r] + bval;
      if (RELU) v = fmaxf(v, 0.0f);
      size_t oidx = ((size_t)(img * a.Cout + ch)) * a.P + ohw;
      if (OUTH)
        ((_Float16*)a.out)[oidx] = (_Float16)v;
      else
        ((float*)a.out)[oidx] = v;
    }
  }
}

// ---------------------------- descriptor builders --------------------------
__global__ void desc_k5(int4* dst, int C, int P, int Kpad) {
  int k = blockIdx.x * blockDim.x + threadIdx.x;
  if (k >= Kpad) return;
  int4 d;
  if (k < C * 25) {
    int c = k / 25, r = k % 25;
    d.x = r / 5 - 2; d.y = r % 5 - 2; d.z = c * P; d.w = 0;
  } else {
    d.x = KBAD; d.y = 0; d.z = 0; d.w = 0;
  }
  dst[k] = d;
}

__global__ void desc_gru(int4* dst, int three, int Kpad) {
  int k = blockIdx.x * blockDim.x + threadIdx.x;
  if (k >= Kpad) return;
  int Keff = three ? 972 : 324;
  int4 d;
  if (k < Keff) {
    int c, e, dil;
    if (three) {
      int dd = k / 324, rem = k % 324;
      c = rem / 9; e = rem % 9;
      dil = (dd == 0) ? 1 : (dd == 1) ? 3 : 5;
    } else {
      c = k / 9; e = k % 9; dil = 1;
    }
    d.x = dil * (e / 3 - 1);
    d.y = dil * (e % 3 - 1);
    d.w = (c >= 18) ? 1 : 0;
    int cc = (c >= 18) ? c - 18 : c;
    d.z = cc * PG;
  } else {
    d.x = KBAD; d.y = 0; d.z = 0; d.w = 0;
  }
  dst[k] = d;
}

__global__ void desc_pac(int4* dst) {
  int k = blockIdx.x * blockDim.x + threadIdx.x;
  if (k >= 800) return;
  int c = k / 25, p = k % 25;
  int4 d;
  d.x = p / 5 - 2; d.y = p % 5 - 2; d.z = c * P0; d.w = p;
  dst[k] = d;
}

// ---------------------------- weight packing -------------------------------
// dst layout per N-tile: [Kpad/2 pairs][16 n][2 parity] f16, tiles contiguous.
__global__ void pack_k5(const float* __restrict__ w, _Float16* __restrict__ dst,
                        int O, int Cin, int Kpad, int iohw) {
  int idx = blockIdx.x * blockDim.x + threadIdx.x;
  int tiles = (O + 15) >> 4;
  int hr = Kpad >> 1;
  if (idx >= tiles * hr * 32) return;
  int t2 = idx & 1;
  int n = (idx >> 1) & 15;
  int rp = idx >> 5;
  int p = rp % hr;
  int tl = rp / hr;
  int k = 2 * p + t2;
  int o = tl * 16 + n;
  float v = 0.0f;
  int K = Cin * 25;
  if (o < O && k < K) {
    if (iohw)  // PAC transpose-conv weights: (in,out,5,5), k=(c,p5) -> w[c][o][p5]
      v = w[((size_t)(k / 25) * O + o) * 25 + (k % 25)];
    else
      v = w[(size_t)o * K + k];
  }
  dst[idx] = (_Float16)v;
}

__global__ void pack_gru1(const float* __restrict__ gg, const float* __restrict__ ga1,
                          const float* __restrict__ ga2, _Float16* __restrict__ dst) {
  int idx = blockIdx.x * blockDim.x + threadIdx.x;
  const int HR = 496;  // Kpad 992 / 2
  if (idx >= 12 * 3 * HR * 32) return;
  int t2 = idx & 1;
  int n = (idx >> 1) & 15;
  int rp = idx >> 5;
  int p = rp % HR;
  int stl = rp / HR;
  int tl = stl % 3;
  int s = stl / 3;
  int k = 2 * p + t2;
  int o = tl * 16 + n;
  float v = 0.0f;
  if (o < 36 && k < 972) {
    int d = k / 324, rem = k % 324, c = rem / 9, e = rem % 9;
    const float* w = (d == 0) ? gg : (d == 1) ? ga1 : ga2;
    v = w[(((size_t)s * 36 + o) * 36 + c) * 9 + e];
  }
  dst[idx] = (_Float16)v;
}

__global__ void pack_gru2(const float* __restrict__ gc, _Float16* __restrict__ dst) {
  int idx = blockIdx.x * blockDim.x + threadIdx.x;
  const int HR = 176;  // Kpad 352 / 2
  if (idx >= 12 * 2 * HR * 32) return;
  int t2 = idx & 1;
  int n = (idx >> 1) & 15;
  int rp = idx >> 5;
  int p = rp % HR;
  int stl = rp / HR;
  int tl = stl % 2;
  int s = stl / 2;
  int k = 2 * p + t2;
  int o = tl * 16 + n;
  float v = 0.0f;
  if (o < 18 && k < 324) {
    int c = k / 9, e = k % 9;
    v = gc[(((size_t)s * 18 + o) * 36 + c) * 9 + e];
  }
  dst[idx] = (_Float16)v;
}

__global__ void bias3(const float* a, const float* b, const float* c, float* d, int n) {
  int i = blockIdx.x * blockDim.x + threadIdx.x;
  if (i < n) d[i] = a[i] + b[i] + c[i];
}

// ---------------------------- small kernels --------------------------------
__global__ void conv_t_k(const float* __restrict__ x, const float* __restrict__ w,
                         const float* __restrict__ bias, float* __restrict__ out) {
  int idx = blockIdx.x * blockDim.x + threadIdx.x;
  if (idx >= BS * 32 * P0) return;
  int hw = idx % P0;
  int o = (idx / P0) % 32;
  int b = idx / (P0 * 32);
  int y = hw / H0, xx = hw % H0;
  float acc = bias[o];
  for (int r = 0; r < 5; ++r)
    for (int s = 0; s < 5; ++s) {
      int iy = y + r - 2, ix = xx + s - 2;
      if (iy >= 0 && iy < H0 && ix >= 0 && ix < H0)
        acc += x[(size_t)b * P0 + iy * H0 + ix] * w[o * 25 + r * 5 + s];
    }
  out[idx] = acc;
}

__global__ void gn_sig(const float* __restrict__ c1, const _Float16* __restrict__ h,
                       const float* __restrict__ gam, const float* __restrict__ bet,
                       _Float16* __restrict__ rh, float* __restrict__ u) {
  __shared__ float ssum[256];
  __shared__ float ssq[256];
  int bid = blockIdx.x;                         // b*12 + t*6 + g
  int b = bid / 12, rest = bid % 12, t = rest / 6, g = rest % 6;
  int chbase = t * 18 + g * 3;
  float s = 0.0f, sq = 0.0f;
  const int N = 3 * PG;
  for (int i = threadIdx.x; i < N; i += 256) {
    int cl = i / PG, hw = i - cl * PG;
    float v = c1[((size_t)(b * 36 + chbase + cl)) * PG + hw];
    s += v; sq += v * v;
  }
  ssum[threadIdx.x] = s; ssq[threadIdx.x] = sq;
  __syncthreads();
  for (int off = 128; off > 0; off >>= 1) {
    if ((int)threadIdx.x < off) {
      ssum[threadIdx.x] += ssum[threadIdx.x + off];
      ssq[threadIdx.x] += ssq[threadIdx.x + off];
    }
    __syncthreads();
  }
  float mean = ssum[0] / N;
  float var = ssq[0] / N - mean * mean;  // biased, like jnp.var
  float rinv = rsqrtf(var + 1e-5f);
  for (int i = threadIdx.x; i < N; i += 256) {
    int cl = i / PG, hw = i - cl * PG;
    int chl = g * 3 + cl;
    float v = c1[((size_t)(b * 36 + chbase + cl)) * PG + hw];
    float yv = (v - mean) * rinv * gam[chl] + bet[chl];
    float sg = 1.0f / (1.0f + __expf(-yv));
    size_t o = ((size_t)(b * 18 + chl)) * PG + hw;
    if (t == 0) rh[o] = (_Float16)(sg * (float)h[o]);   // r * h
    else        u[o] = sg;                              // update gate
  }
}

__global__ void gru_update(const float* __restrict__ ct, const float* __restrict__ u,
                           const _Float16* __restrict__ h, _Float16* __restrict__ hn,
                           int n) {
  int i = blockIdx.x * blockDim.x + threadIdx.x;
  if (i >= n) return;
  float uu = u[i];
  hn[i] = (_Float16)(uu * (float)h[i] + (1.0f - uu) * tanhf(ct[i]));
}

__global__ void pac_affinity(const _Float16* __restrict__ gf, float* __restrict__ Kout) {
  int idx = blockIdx.x * blockDim.x + threadIdx.x;
  if (idx >= BS * PG) return;
  int b = idx / PG, hw = idx % PG;
  int y = hw / HG, x = hw % HG;
  float ctr[18];
#pragma unroll
  for (int c = 0; c < 18; ++c)
    ctr[c] = (float)gf[((size_t)(b * 18 + c)) * PG + hw];
  for (int p = 0; p < 25; ++p) {
    int iy = y + p / 5 - 2, ix = x + p % 5 - 2;
    bool inb = (iy >= 0 && iy < HG && ix >= 0 && ix < HG);
    float s = 0.0f;
#pragma unroll
    for (int c = 0; c < 18; ++c) {
      float v = inb ? (float)gf[((size_t)(b * 18 + c) * HG + iy) * HG + ix] : 0.0f;
      float d = v - ctr[c];
      s += d * d;
    }
    Kout[((size_t)(b * 25 + p)) * PG + hw] = __expf(-0.5f * s);
  }
}

// ---------------------------- launcher -------------------------------------
extern "C" void kernel_launch(void* const* d_in, const int* in_sizes, int n_in,
                              void* d_out, int out_size, void* d_ws, size_t ws_size,
                              hipStream_t stream) {
  (void)in_sizes; (void)n_in; (void)out_size; (void)ws_size;
  const float* x_in  = (const float*)d_in[0];
  const float* guide = (const float*)d_in[1];
  const float* t_w   = (const float*)d_in[2];
  const float* t_b   = (const float*)d_in[3];
  const float* g1_w  = (const float*)d_in[4];
  const float* g1_b  = (const float*)d_in[5];
  const float* g2_w  = (const float*)d_in[6];
  const float* g2_b  = (const float*)d_in[7];
  const float* g3_w  = (const float*)d_in[8];
  const float* g3_b  = (const float*)d_in[9];
  const float* gg_w  = (const float*)d_in[10];
  const float* gg_b  = (const float*)d_in[11];
  const float* ga1_w = (const float*)d_in[12];
  const float* ga1_b = (const float*)d_in[13];
  const float* ga2_w = (const float*)d_in[14];
  const float* ga2_b = (const float*)d_in[15];
  const float* gc_w  = (const float*)d_in[16];
  const float* gc_b  = (const float*)d_in[17];
  const float* gn_g  = (const float*)d_in[18];
  const float* gn_b  = (const float*)d_in[19];
  const float* up_w  = (const float*)d_in[20];
  const float* up_b  = (const float*)d_in[21];
  const float* f1_w  = (const float*)d_in[22];
  const float* f1_b  = (const float*)d_in[23];
  const float* f2_w  = (const float*)d_in[24];
  const float* f2_b  = (const float*)d_in[25];

  char* ws = (char*)d_ws;
  size_t off = 0;
  auto alloc = [&](size_t bytes) -> char* {
    char* p = ws + off;
    off += (bytes + 255) & ~(size_t)255;
    return p;
  };

  _Float16* pk_g1  = (_Float16*)alloc((size_t)2 * 10 * 1024);
  _Float16* pk_g2  = (_Float16*)alloc((size_t)2 * 25 * 1024);
  _Float16* pk_g3  = (_Float16*)alloc((size_t)2 * 25 * 1024);
  _Float16* pk_f1  = (_Float16*)alloc((size_t)2 * 25 * 1024);
  _Float16* pk_f2  = (_Float16*)alloc((size_t)1 * 25 * 1024);
  _Float16* pk_up  = (_Float16*)alloc((size_t)2 * 25 * 1024);
  _Float16* pk_gr1 = (_Float16*)alloc((size_t)12 * 3 * 31 * 1024);
  _Float16* pk_gr2 = (_Float16*)alloc((size_t)12 * 2 * 11 * 1024);
  int4* ds_k5a = (int4*)alloc((size_t)320 * 16);
  int4* ds_k5b = (int4*)alloc((size_t)800 * 16);
  int4* ds_gr1 = (int4*)alloc((size_t)992 * 16);
  int4* ds_gr2 = (int4*)alloc((size_t)352 * 16);
  int4* ds_pac = (int4*)alloc((size_t)800 * 16);
  float*    bias_c1 = (float*)alloc((size_t)12 * 36 * 4);
  float*    xt  = (float*)alloc((size_t)BS * 32 * P0 * 4);
  _Float16* g3h = (_Float16*)alloc((size_t)24 * 18 * PG * 2);
  char* regionB = alloc((size_t)2 * 24 * 32 * PG * 2);  // reused region

  _Float16* g1h = (_Float16*)regionB;
  _Float16* g2h = (_Float16*)(regionB + (size_t)24 * 32 * PG * 2);
  size_t c2 = 0;
  auto sub = [&](size_t bytes) -> char* {
    char* p = regionB + c2;
    c2 += (bytes + 255) & ~(size_t)255;
    return p;
  };
  _Float16* h0    = (_Float16*)sub((size_t)BS * 18 * PG * 2);
  _Float16* h1    = (_Float16*)sub((size_t)BS * 18 * PG * 2);
  _Float16* rh    = (_Float16*)sub((size_t)BS * 18 * PG * 2);
  float*    ubuf  = (float*)sub((size_t)BS * 18 * PG * 4);
  float*    c1buf = (float*)sub((size_t)BS * 36 * PG * 4);
  float*    ctbuf = (float*)sub((size_t)BS * 18 * PG * 4);
  float*    Kmap  = (float*)sub((size_t)BS * 25 * PG * 4);
  _Float16* xu    = (_Float16*)sub((size_t)BS * 32 * PG * 2);
  _Float16* xf1   = (_Float16*)sub((size_t)BS * 32 * PG * 2);

  // ---- pack weights + descriptors ----
  pack_k5<<<dim3((2 * 160 * 32 + 255) / 256), 256, 0, stream>>>(g1_w, pk_g1, 32, 12, 320, 0);
  pack_k5<<<dim3((2 * 400 * 32 + 255) / 256), 256, 0, stream>>>(g2_w, pk_g2, 32, 32, 800, 0);
  pack_k5<<<dim3((2 * 400 * 32 + 255) / 256), 256, 0, stream>>>(g3_w, pk_g3, 18, 32, 800, 0);
  pack_k5<<<dim3((2 * 400 * 32 + 255) / 256), 256, 0, stream>>>(f1_w, pk_f1, 32, 32, 800, 0);
  pack_k5<<<dim3((1 * 400 * 32 + 255) / 256), 256, 0, stream>>>(f2_w, pk_f2, 1, 32, 800, 0);
  pack_k5<<<dim3((2 * 400 * 32 + 255) / 256), 256, 0, stream>>>(up_w, pk_up, 32, 32, 800, 1);
  pack_gru1<<<dim3((12 * 3 * 496 * 32 + 255) / 256), 256, 0, stream>>>(gg_w, ga1_w, ga2_w, pk_gr1);
  pack_gru2<<<dim3((12 * 2 * 176 * 32 + 255) / 256), 256, 0, stream>>>(gc_w, pk_gr2);
  desc_k5<<<dim3(2), 256, 0, stream>>>(ds_k5a, 12, PG, 320);
  desc_k5<<<dim3(4), 256, 0, stream>>>(ds_k5b, 32, PG, 800);
  desc_gru<<<dim3(4), 256, 0, stream>>>(ds_gr1, 1, 992);
  desc_gru<<<dim3(2), 256, 0, stream>>>(ds_gr2, 0, 352);
  desc_pac<<<dim3(4), 256, 0, stream>>>(ds_pac);
  bias3<<<dim3(2), 256, 0, stream>>>(gg_b, ga1_b, ga2_b, bias_c1, 12 * 36);

  // ---- branch_t (tiny, scalar) ----
  conv_t_k<<<dim3(BS * 32 * P0 / 256), 256, 0, stream>>>(x_in, t_w, t_b, xt);

  // ---- guide branch, WMMA implicit GEMM ----
  ConvArgs a{};
  a.in0 = guide; a.in1 = nullptr; a.kmap = nullptr;
  a.bpack = pk_g1; a.desc = ds_k5a; a.bias = g1_b; a.out = g1h;
  a.H = HG; a.W = HG; a.P = PG; a.Cout = 32;
  a.Kchunks = 10; a.stride0 = 12 * PG;
  wmma_conv<MK5F32, true, true><<<dim3(24 * PG / 64, 2), 128, 10 * 1536, stream>>>(a);

  a.in0 = g1h; a.bpack = pk_g2; a.desc = ds_k5b; a.bias = g2_b; a.out = g2h;
  a.Kchunks = 25; a.stride0 = 32 * PG;
  wmma_conv<MK5F16, true, true><<<dim3(24 * PG / 64, 2), 128, 25 * 1536, stream>>>(a);

  a.in0 = g2h; a.bpack = pk_g3; a.bias = g3_b; a.out = g3h; a.Cout = 18;
  wmma_conv<MK5F16, false, true><<<dim3(24 * PG / 64, 2), 128, 25 * 1536, stream>>>(a);

  // ---- ConvGRU, 12 steps ----
  hipMemsetAsync(h0, 0, (size_t)BS * 18 * PG * 2, stream);
  _Float16* hc = h0;
  _Float16* hn = h1;
  for (int s = 0; s < 12; ++s) {
    const _Float16* gi = g3h + (size_t)s * BS * 18 * PG;
    // fused 3-dilation 36->36 conv (K = 3*36*9 = 972)
    a.in0 = gi; a.in1 = hc; a.kmap = nullptr;
    a.bpack = pk_gr1 + (size_t)s * 3 * 496 * 32; a.desc = ds_gr1;
    a.bias = bias_c1 + s * 36; a.out = c1buf;
    a.Cout = 36; a.Kchunks = 31; a.stride0 = 18 * PG;
    wmma_conv<MGRU, false, false><<<dim3(BS * PG / 64, 3), 128, 31 * 1536, stream>>>(a);

    gn_sig<<<dim3(24), 256, 0, stream>>>(c1buf, hc, gn_g + s * 18, gn_b + s * 18, rh, ubuf);

    // candidate conv 36->18 (K = 36*9 = 324)
    a.in1 = rh; a.bpack = pk_gr2 + (size_t)s * 2 * 176 * 32; a.desc = ds_gr2;
    a.bias = gc_b + s * 18; a.out = ctbuf;
    a.Cout = 18; a.Kchunks = 11;
    wmma_conv<MGRU, false, false><<<dim3(BS * PG / 64, 2), 128, 11 * 1536, stream>>>(a);

    gru_update<<<dim3((BS * 18 * PG + 255) / 256), 256, 0, stream>>>(
        ctbuf, ubuf, hc, hn, BS * 18 * PG);
    _Float16* tmp = hc; hc = hn; hn = tmp;
  }

  // ---- PAC affinity + pixel-adaptive transpose conv ----
  pac_affinity<<<dim3((BS * PG + 127) / 128), 128, 0, stream>>>(hc, Kmap);

  a.in0 = xt; a.in1 = nullptr; a.kmap = Kmap;
  a.bpack = pk_up; a.desc = ds_pac; a.bias = up_b; a.out = xu;
  a.Cout = 32; a.Kchunks = 25; a.stride0 = 32 * P0;
  wmma_conv<MPAC, true, true><<<dim3(BS * PG / 64, 2), 128, 25 * 1536, stream>>>(a);

  // ---- branch_f ----
  a.in0 = xu; a.kmap = nullptr; a.bpack = pk_f1; a.desc = ds_k5b;
  a.bias = f1_b; a.out = xf1; a.stride0 = 32 * PG;
  wmma_conv<MK5F16, true, true><<<dim3(BS * PG / 64, 2), 128, 25 * 1536, stream>>>(a);

  a.in0 = xf1; a.bpack = pk_f2; a.bias = f2_b; a.out = d_out; a.Cout = 1;
  wmma_conv<MK5F16, false, false><<<dim3(BS * PG / 64, 1), 128, 25 * 1536, stream>>>(a);
}
